// GNNmodel_42228118454876
// MI455X (gfx1250) — matrix operand order repaired and checked by reading, not compiled, not run on hardware
//
#include <hip/hip_runtime.h>
#include <hip/hip_bf16.h>

typedef __attribute__((ext_vector_type(16))) _Float16 v16h;
typedef __attribute__((ext_vector_type(8)))  float    v8f;

#define N_NODES   50000
#define N_EDGES   800000
#define IN_FEATS  128
#define NUM_HEADS 4
#define OUT_FEATS 32
#define HF        128
#define NEG_SLOPE 0.2f

// -------- monotone float<->uint key for atomicMax-based float max --------
__device__ __forceinline__ unsigned fkey(float f) {
    unsigned u = __float_as_uint(f);
    return (u >> 31) ? ~u : (u | 0x80000000u);
}
__device__ __forceinline__ float funkey(unsigned k) {
    return (k >> 31) ? __uint_as_float(k & 0x7fffffffu) : __uint_as_float(~k);
}

// -------- init: zero rst accumulator, emax keys, denom --------
__global__ void k_init(float* __restrict__ rst, unsigned* __restrict__ emax,
                       float* __restrict__ denom) {
    int i = blockIdx.x * blockDim.x + threadIdx.x;
    int stride = gridDim.x * blockDim.x;
    for (int j = i; j < N_NODES * HF; j += stride) rst[j] = 0.0f;
    for (int j = i; j < N_NODES * NUM_HEADS; j += stride) { emax[j] = 0u; denom[j] = 0.0f; }
}

// -------- fused dual GEMM on WMMA: h = feat@W_fc, idn(out) = feat@W_idn --------
// Block = 256 threads = 8 waves. Block handles a 16-row node tile; wave w
// handles output columns [16w, 16w+16). A-fragment shared between both GEMMs.
__global__ __launch_bounds__(256) void k_gemm(const float* __restrict__ feat,
                                              const float* __restrict__ Wfc,
                                              const float* __restrict__ Widn,
                                              float* __restrict__ h,
                                              float* __restrict__ idn) {
    const int m0   = blockIdx.x * 16;
    const int wave = threadIdx.x >> 5;
    const int lane = threadIdx.x & 31;
    const int n0   = wave * 16;
    const int mrow = m0 + (lane & 15);
    const int hi   = lane >> 4;

    v8f c_fc = {};
    v8f c_id = {};

    for (int kc = 0; kc < IN_FEATS; kc += 32) {
        // A fragment: 16x32 f16 (ISA 7.12.2 layout)
        const float* arow = feat + mrow * IN_FEATS + kc;
        v16h a;
#pragma unroll
        for (int i = 0; i < 8; ++i) {
            a[i]     = (_Float16)arow[hi * 8 + i];
            a[i + 8] = (_Float16)arow[16 + hi * 8 + i];
        }
        // B fragments: 32x16 f16, lane&15 = N, elements = K in same order as A
        const float* b0 = Wfc  + kc * HF + n0 + (lane & 15);
        const float* b1 = Widn + kc * HF + n0 + (lane & 15);
        v16h bfc, bid;
#pragma unroll
        for (int i = 0; i < 8; ++i) {
            bfc[i]     = (_Float16)b0[(hi * 8 + i) * HF];
            bfc[i + 8] = (_Float16)b0[(16 + hi * 8 + i) * HF];
            bid[i]     = (_Float16)b1[(hi * 8 + i) * HF];
            bid[i + 8] = (_Float16)b1[(16 + hi * 8 + i) * HF];
        }
        c_fc = __builtin_amdgcn_wmma_f32_16x16x32_f16(false, a, false, bfc,
                                                      (short)0, c_fc, false, false);
        c_id = __builtin_amdgcn_wmma_f32_16x16x32_f16(false, a, false, bid,
                                                      (short)0, c_id, false, false);
    }

    // C/D layout: lane&15 = column, VGPR v = row 8*(lane>>4)+v
    const int col   = n0 + (lane & 15);
    const int rbase = m0 + hi * 8;
#pragma unroll
    for (int v = 0; v < 8; ++v) {
        h[(rbase + v) * HF + col]   = c_fc[v];
        idn[(rbase + v) * HF + col] = c_id[v];
    }
}

// -------- per (node, head) attention halves: el = h.attn_l, er = h.attn_r --------
__global__ void k_attn(const float* __restrict__ h, const float* __restrict__ attn_l,
                       const float* __restrict__ attn_r, float* __restrict__ el,
                       float* __restrict__ er) {
    int idx = blockIdx.x * blockDim.x + threadIdx.x;
    if (idx >= N_NODES * NUM_HEADS) return;
    int n = idx >> 2, hd = idx & 3;
    const float* hp = h + n * HF + hd * OUT_FEATS;
    const float* al = attn_l + hd * OUT_FEATS;
    const float* ar = attn_r + hd * OUT_FEATS;
    float sl = 0.0f, sr = 0.0f;
#pragma unroll
    for (int f = 0; f < OUT_FEATS; ++f) {
        float v = hp[f];
        sl += v * al[f];
        sr += v * ar[f];
    }
    el[idx] = sl;
    er[idx] = sr;
}

__device__ __forceinline__ float edge_logit(const float* el, const float* er,
                                            int s, int d, int hd) {
    float z = el[s * NUM_HEADS + hd] + er[d * NUM_HEADS + hd];
    return z > 0.0f ? z : NEG_SLOPE * z;
}

// -------- segment max over destination neighborhoods --------
__global__ void k_emax(const int* __restrict__ src, const int* __restrict__ dst,
                       const float* __restrict__ el, const float* __restrict__ er,
                       unsigned* __restrict__ emax) {
    int idx = blockIdx.x * blockDim.x + threadIdx.x;
    if (idx >= N_EDGES * NUM_HEADS) return;
    int e = idx >> 2, hd = idx & 3;
    int s = src[e], d = dst[e];
    float z = edge_logit(el, er, s, d, hd);
    atomicMax(emax + d * NUM_HEADS + hd, fkey(z));
}

// -------- segment sum of exp(logit - max) --------
__global__ void k_denom(const int* __restrict__ src, const int* __restrict__ dst,
                        const float* __restrict__ el, const float* __restrict__ er,
                        const unsigned* __restrict__ emax, float* __restrict__ denom) {
    int idx = blockIdx.x * blockDim.x + threadIdx.x;
    if (idx >= N_EDGES * NUM_HEADS) return;
    int e = idx >> 2, hd = idx & 3;
    int s = src[e], d = dst[e];
    float z  = edge_logit(el, er, s, d, hd);
    float ex = __expf(z - funkey(emax[d * NUM_HEADS + hd]));
    atomicAdd(denom + d * NUM_HEADS + hd, ex);
}

// -------- wave-per-edge attention-weighted scatter: rst[dst] += a * h[src] --------
__global__ __launch_bounds__(256) void k_scatter(const int* __restrict__ src,
                                                 const int* __restrict__ dst,
                                                 const float* __restrict__ el,
                                                 const float* __restrict__ er,
                                                 const unsigned* __restrict__ emax,
                                                 const float* __restrict__ denom,
                                                 const float* __restrict__ h,
                                                 float* __restrict__ rst) {
    int e = blockIdx.x * (blockDim.x >> 5) + (threadIdx.x >> 5);
    if (e >= N_EDGES) return;
    int lane = threadIdx.x & 31;
    int hd   = lane >> 3;                      // 8 lanes per head (32 feats / 4 per lane)
    int s = src[e], d = dst[e];
    float z  = edge_logit(el, er, s, d, hd);
    float ex = __expf(z - funkey(emax[d * NUM_HEADS + hd]));
    float a  = ex / denom[d * NUM_HEADS + hd];
    const float4 hv = *reinterpret_cast<const float4*>(h + s * HF + lane * 4);
    float* rp = rst + d * HF + lane * 4;
    atomicAdd(rp + 0, hv.x * a);
    atomicAdd(rp + 1, hv.y * a);
    atomicAdd(rp + 2, hv.z * a);
    atomicAdd(rp + 3, hv.w * a);
}

// -------- wave-per-node SE block + residual: out = rst * sigmoid(...) + idn --------
__global__ __launch_bounds__(256) void k_final(const float* __restrict__ rst,
                                               const float* __restrict__ se_w1,
                                               const float* __restrict__ se_w2,
                                               float* __restrict__ out) {
    int n = blockIdx.x * (blockDim.x >> 5) + (threadIdx.x >> 5);
    if (n >= N_NODES) return;
    int lane = threadIdx.x & 31;
    int hd   = lane >> 3;
    const float4 rv = *reinterpret_cast<const float4*>(rst + n * HF + lane * 4);
    float s = rv.x + rv.y + rv.z + rv.w;
    // reduce over the 8 lanes of this head (wave32 shuffles)
    s += __shfl_xor(s, 1, 32);
    s += __shfl_xor(s, 2, 32);
    s += __shfl_xor(s, 4, 32);
    float ymean = s * (1.0f / OUT_FEATS);
    // squeeze: t = relu(sum_h ymean[h] * se_w1[h])  (SE_HIDDEN == 1)
    float t = 0.0f;
#pragma unroll
    for (int hh = 0; hh < NUM_HEADS; ++hh)
        t += __shfl(ymean, hh * 8, 32) * se_w1[hh];
    t = t > 0.0f ? t : 0.0f;
    // excite + sigmoid, per head
    float yv = 1.0f / (1.0f + __expf(-(t * se_w2[hd])));
    float4 ov = *reinterpret_cast<float4*>(out + n * HF + lane * 4);  // holds idn
    ov.x = ov.x + rv.x * yv;
    ov.y = ov.y + rv.y * yv;
    ov.z = ov.z + rv.z * yv;
    ov.w = ov.w + rv.w * yv;
    *reinterpret_cast<float4*>(out + n * HF + lane * 4) = ov;
}

extern "C" void kernel_launch(void* const* d_in, const int* in_sizes, int n_in,
                              void* d_out, int out_size, void* d_ws, size_t ws_size,
                              hipStream_t stream) {
    (void)in_sizes; (void)n_in; (void)out_size; (void)ws_size;
    const float* feat   = (const float*)d_in[0];
    const float* Wfc    = (const float*)d_in[1];
    const float* attn_l = (const float*)d_in[2];
    const float* attn_r = (const float*)d_in[3];
    const float* se_w1  = (const float*)d_in[4];
    const float* se_w2  = (const float*)d_in[5];
    const float* Widn   = (const float*)d_in[6];
    const int*   src    = (const int*)d_in[7];
    const int*   dst    = (const int*)d_in[8];
    float* out = (float*)d_out;

    // workspace layout (bytes, all 16B aligned)
    char* ws = (char*)d_ws;
    float*    h     = (float*)(ws);                    // 50000*128*4 = 25,600,000
    float*    rst   = (float*)(ws + 25600000);         // 25,600,000
    float*    el    = (float*)(ws + 51200000);         //    800,000
    float*    er    = (float*)(ws + 52000000);         //    800,000
    unsigned* emax  = (unsigned*)(ws + 52800000);      //    800,000
    float*    denom = (float*)(ws + 53600000);         //    800,000  (end: 54.4 MB)

    k_init<<<512, 256, 0, stream>>>(rst, emax, denom);
    k_gemm<<<N_NODES / 16, 256, 0, stream>>>(feat, Wfc, Widn, h, out);
    k_attn<<<(N_NODES * NUM_HEADS + 255) / 256, 256, 0, stream>>>(h, attn_l, attn_r, el, er);
    k_emax<<<(N_EDGES * NUM_HEADS + 255) / 256, 256, 0, stream>>>(src, dst, el, er, emax);
    k_denom<<<(N_EDGES * NUM_HEADS + 255) / 256, 256, 0, stream>>>(src, dst, el, er, emax, denom);
    k_scatter<<<(N_EDGES + 7) / 8, 256, 0, stream>>>(src, dst, el, er, emax, denom, h, rst);
    k_final<<<(N_NODES + 7) / 8, 256, 0, stream>>>(rst, se_w1, se_w2, out);
}
